// MultiHeadAttentionClassical_65481071401116
// MI455X (gfx1250) — matrix operand order repaired and checked by reading, not compiled, not run on hardware
//
#include <hip/hip_runtime.h>
#include <hip/hip_bf16.h>

typedef unsigned short u16;
typedef unsigned int   u32;
typedef __attribute__((ext_vector_type(16))) __bf16 v16bf;
typedef __attribute__((ext_vector_type(8)))  float  v8f;
typedef __attribute__((ext_vector_type(8)))  u32    v8u;

#define EMBED 1024
#define SEQ   2048
#define BATCH 2
#define HEADS 16
#define DK    64
#define MROWS (BATCH * SEQ)   // 4096

// ---------- CDNA5 async global->LDS path (guarded) ----------

#if defined(__AMDGCN__) && __has_builtin(__builtin_amdgcn_global_load_async_to_lds_b128)
#define HAVE_ASYNC 1
#else
#define HAVE_ASYNC 0
#endif

__device__ __forceinline__ void cp16(u16* ldst, const u16* gsrc) {
#if HAVE_ASYNC
  typedef int i4 __attribute__((vector_size(16)));     // matches builtin's V4i param
  typedef __attribute__((address_space(1))) i4 gi4;
  typedef __attribute__((address_space(3))) i4 li4;
  // LDS generic pointer: low 32 bits are the LDS byte offset (aperture map).
  gi4* g = (gi4*)(unsigned long long)gsrc;
  li4* l = (li4*)(unsigned int)(unsigned long long)ldst;
  __builtin_amdgcn_global_load_async_to_lds_b128(g, l, 0, 0);
#else
  *(uint4*)ldst = *(const uint4*)gsrc;
#endif
}

__device__ __forceinline__ void wait_async() {
#if HAVE_ASYNC
#if __has_builtin(__builtin_amdgcn_s_wait_asynccnt)
  __builtin_amdgcn_s_wait_asynccnt(0);
#else
  asm volatile("s_wait_asynccnt 0x0" ::: "memory");
#endif
#endif
}

// ---------- helpers ----------

__device__ __forceinline__ u16 f2bf(float f) {
  u32 u = __builtin_bit_cast(u32, f);
  u += 0x7fffu + ((u >> 16) & 1u);      // round-to-nearest-even
  return (u16)(u >> 16);
}

__device__ __forceinline__ u32 pk2(float lo, float hi) {
  return (u32)f2bf(lo) | ((u32)f2bf(hi) << 16);
}

__device__ __forceinline__ float rmax16(float v) {
#pragma unroll
  for (int m = 8; m >= 1; m >>= 1) v = fmaxf(v, __shfl_xor(v, m, 32));
  return v;
}

__device__ __forceinline__ float rsum16(float v) {
#pragma unroll
  for (int m = 8; m >= 1; m >>= 1) v += __shfl_xor(v, m, 32);
  return v;
}

// A fragment (16x32 bf16, row-major in LDS, K contiguous).
// lanes 0-15: row M=lane, K {0..7,16..23}; lanes 16-31: K {8..15,24..31}.
__device__ __forceinline__ v16bf lds_frag_a(const u16* s, int row0, int k0, int pitch) {
  int lane = threadIdx.x & 31;
  int r = lane & 15, h = lane >> 4;
  const u16* p = s + (size_t)(row0 + r) * pitch + k0 + h * 8;
  v8u raw;
#pragma unroll
  for (int j = 0; j < 4; ++j) raw[j]     = *(const u32*)(p + 2 * j);
#pragma unroll
  for (int j = 0; j < 4; ++j) raw[j + 4] = *(const u32*)(p + 16 + 2 * j);
  return __builtin_bit_cast(v16bf, raw);
}

// B fragment (32x16 bf16): element (k,n) = S[n][k], S row-major N x K in LDS.
// lane holds column n=lane%16; lanes 0-15 K {0..7,16..23}, lanes 16-31 K {8..15,24..31}.
__device__ __forceinline__ v16bf lds_frag_b(const u16* s, int n0, int k0, int pitch) {
  int lane = threadIdx.x & 31;
  int r = lane & 15, h = lane >> 4;
  const u16* p = s + (size_t)(n0 + r) * pitch + k0 + h * 8;
  v8u raw;
#pragma unroll
  for (int j = 0; j < 4; ++j) raw[j]     = *(const u32*)(p + 2 * j);
#pragma unroll
  for (int j = 0; j < 4; ++j) raw[j + 4] = *(const u32*)(p + 16 + 2 * j);
  return __builtin_bit_cast(v16bf, raw);
}

__device__ __forceinline__ v8f wmma_bf16(v16bf a, v16bf b, v8f c) {
  return __builtin_amdgcn_wmma_f32_16x16x32_bf16(false, a, false, b, (short)0, c,
                                                 false, false);
}

// ---------- fp32 -> bf16 conversion ----------

__global__ __launch_bounds__(256) void f32_to_bf16_kernel(
    const float* __restrict__ src, u16* __restrict__ dst, int n8) {
  int i = blockIdx.x * 256 + threadIdx.x;
  if (i >= n8) return;
  const float4* s = (const float4*)src;
  float4 a = s[2 * i], b = s[2 * i + 1];
  uint4 o;
  o.x = pk2(a.x, a.y);
  o.y = pk2(a.z, a.w);
  o.z = pk2(b.x, b.y);
  o.w = pk2(b.z, b.w);
  ((uint4*)dst)[i] = o;
}

// ---------- tiled WMMA GEMM: y = A(MxK) @ W(NxK)^T ----------
// mode 0: bf16 out, head-major [b,h,s,d]            (Q, K projections)
// mode 2: bf16 out, head-major transposed [b,h,d,s] (V projection -> V^T)
// mode 3: f32 out, row-major [M,N]                  (output projection)

#define TKC 64  // K-chunk
#define GP  72  // LDS pitch (ushorts): 144B rows -> 16B aligned, conflict-free

__global__ __launch_bounds__(256) void gemm_bf16_kernel(
    const u16* __restrict__ A, const u16* __restrict__ W,
    u16* __restrict__ outb, float* __restrict__ outf, int mode) {
  __shared__ u16 sA[2][128 * GP];
  __shared__ u16 sB[2][64 * GP];

  const int tid = threadIdx.x;
  const int m0 = blockIdx.x * 128;
  const int n0 = blockIdx.y * 64;
  const int w  = tid >> 5;
  const int wm = (w & 3) * 32;
  const int wn = (w >> 2) * 32;

  auto stage = [&](int buf, int kc) {
#pragma unroll
    for (int t = 0; t < 4; ++t) {           // A tile 128x64: 1024 x 16B
      int q = tid + t * 256;
      int row = q >> 3, c4 = q & 7;
      cp16(&sA[buf][row * GP + c4 * 8],
           A + (size_t)(m0 + row) * EMBED + kc + c4 * 8);
    }
#pragma unroll
    for (int t = 0; t < 2; ++t) {           // B tile 64x64: 512 x 16B
      int q = tid + t * 256;
      int row = q >> 3, c4 = q & 7;
      cp16(&sB[buf][row * GP + c4 * 8],
           W + (size_t)(n0 + row) * EMBED + kc + c4 * 8);
    }
  };

  v8f c00 = {}, c01 = {}, c10 = {}, c11 = {};

  stage(0, 0);
  wait_async();
  __syncthreads();

  int cur = 0;
  for (int kc = 0; kc < EMBED; kc += TKC) {
    if (kc + TKC < EMBED) {
      stage(cur ^ 1, kc + TKC);             // async: overlaps with WMMAs below
      if (kc + 2 * TKC < EMBED)
        __builtin_prefetch(A + (size_t)(m0 + (tid >> 1)) * EMBED + kc + 2 * TKC, 0, 1);
    }
    const u16* pA = sA[cur];
    const u16* pB = sB[cur];
#pragma unroll
    for (int kh = 0; kh < TKC; kh += 32) {
      v16bf a0 = lds_frag_a(pA, wm, kh, GP);
      v16bf a1 = lds_frag_a(pA, wm + 16, kh, GP);
      v16bf b0 = lds_frag_b(pB, wn, kh, GP);
      v16bf b1 = lds_frag_b(pB, wn + 16, kh, GP);
      c00 = wmma_bf16(a0, b0, c00);
      c01 = wmma_bf16(a0, b1, c01);
      c10 = wmma_bf16(a1, b0, c10);
      c11 = wmma_bf16(a1, b1, c11);
    }
    wait_async();
    __syncthreads();
    cur ^= 1;
  }

  const int lane = tid & 31;
  const int nl = lane & 15, hh = lane >> 4;
  v8f cf[2][2] = {{c00, c01}, {c10, c11}};
#pragma unroll
  for (int i = 0; i < 2; ++i)
#pragma unroll
    for (int j = 0; j < 2; ++j)
#pragma unroll
      for (int vi = 0; vi < 8; ++vi) {
        int m = m0 + wm + i * 16 + vi + 8 * hh;
        int n = n0 + wn + j * 16 + nl;
        float v = cf[i][j][vi];
        if (mode == 3) {
          outf[(size_t)m * EMBED + n] = v;
        } else {
          int b = m >> 11, sIdx = m & (SEQ - 1);
          int hd = n >> 6, d = n & (DK - 1);
          size_t idx;
          if (mode == 2)
            idx = ((size_t)(b * HEADS + hd) * DK + d) * SEQ + sIdx;   // V^T
          else
            idx = ((size_t)(b * HEADS + hd) * SEQ + sIdx) * DK + d;   // Q/K
          outb[idx] = f2bf(v);
        }
      }
}

// ---------- flash attention: softmax(Q K^T / 8) V per (b,h) ----------

#define AP 72   // LDS pitch for all attention tiles

__global__ __launch_bounds__(128) void attn_kernel(
    const u16* __restrict__ Q, const u16* __restrict__ K,
    const u16* __restrict__ VT, u16* __restrict__ O) {
  __shared__ u16 sQ[64 * AP];
  __shared__ u16 sK[2][64 * AP];
  __shared__ u16 sV[2][64 * AP];
  __shared__ u16 sP[4][16 * AP];   // per-wave P scratch (16 x 64)

  const int tid  = threadIdx.x;
  const int w    = tid >> 5;
  const int lane = tid & 31;
  const int nl = lane & 15, hh = lane >> 4;
  const int bh = blockIdx.y;           // 0..31
  const int q0 = blockIdx.x * 64;

  const size_t qkBase = (size_t)bh * SEQ * DK;
  const size_t vBase  = (size_t)bh * DK * SEQ;

  auto stageKV = [&](int buf, int kv) {
#pragma unroll
    for (int t = 0; t < 4; ++t) {     // 64x64 each: 512 x 16B per tile
      int q = tid + t * 128;
      int row = q >> 3, c4 = q & 7;
      cp16(&sK[buf][row * AP + c4 * 8],
           K + qkBase + (size_t)(kv + row) * DK + c4 * 8);
      cp16(&sV[buf][row * AP + c4 * 8],
           VT + vBase + (size_t)row * SEQ + kv + c4 * 8);
    }
  };

  // stage Q tile 64x64 + first K/V tiles
#pragma unroll
  for (int t = 0; t < 4; ++t) {
    int q = tid + t * 128;
    int row = q >> 3, c4 = q & 7;
    cp16(&sQ[row * AP + c4 * 8],
         Q + qkBase + (size_t)(q0 + row) * DK + c4 * 8);
  }
  stageKV(0, 0);
  wait_async();
  __syncthreads();

  v16bf qa0 = lds_frag_a(sQ, w * 16, 0, AP);
  v16bf qa1 = lds_frag_a(sQ, w * 16, 32, AP);

  v8f o0 = {}, o1 = {}, o2 = {}, o3 = {};
  float mrow[8], lrow[8];
#pragma unroll
  for (int i = 0; i < 8; ++i) { mrow[i] = -1e30f; lrow[i] = 0.f; }

  u16* sPw = sP[w];
  int cur = 0;

  for (int kv = 0; kv < SEQ; kv += 64) {
    if (kv + 64 < SEQ) stageKV(cur ^ 1, kv + 64);  // async overlap

    const u16* pK = sK[cur];
    const u16* pV = sV[cur];

    // scores: 16 q-rows x 64 keys (4 key-groups), K-dim = d (two 32-chunks)
    v8f s[4];
#pragma unroll
    for (int g = 0; g < 4; ++g) {
      s[g] = (v8f){};
      s[g] = wmma_bf16(qa0, lds_frag_b(pK, g * 16, 0, AP), s[g]);
      s[g] = wmma_bf16(qa1, lds_frag_b(pK, g * 16, 32, AP), s[g]);
#pragma unroll
      for (int i = 0; i < 8; ++i) s[g][i] *= 0.125f;
    }

    // online softmax (row stats replicated across each 16-lane half)
#pragma unroll
    for (int i = 0; i < 8; ++i) {
      float rm = rmax16(fmaxf(fmaxf(s[0][i], s[1][i]), fmaxf(s[2][i], s[3][i])));
      float mn = fmaxf(mrow[i], rm);
      float alpha = __expf(mrow[i] - mn);
      float p0 = __expf(s[0][i] - mn);
      float p1 = __expf(s[1][i] - mn);
      float p2 = __expf(s[2][i] - mn);
      float p3 = __expf(s[3][i] - mn);
      s[0][i] = p0; s[1][i] = p1; s[2][i] = p2; s[3][i] = p3;
      lrow[i] = lrow[i] * alpha + rsum16(p0 + p1 + p2 + p3);
      mrow[i] = mn;
      o0[i] *= alpha; o1[i] *= alpha; o2[i] *= alpha; o3[i] *= alpha;
    }

    // C-layout -> A-layout transpose of P (16x64) through wave-private LDS
#pragma unroll
    for (int i = 0; i < 8; ++i) {
      int m = i + 8 * hh;
#pragma unroll
      for (int g = 0; g < 4; ++g)
        sPw[m * AP + g * 16 + nl] = f2bf(s[g][i]);
    }

    v16bf pa0 = lds_frag_a(sPw, 0, 0, AP);
    v16bf pa1 = lds_frag_a(sPw, 0, 32, AP);
    o0 = wmma_bf16(pa0, lds_frag_b(pV, 0,  0, AP), o0);
    o1 = wmma_bf16(pa0, lds_frag_b(pV, 16, 0, AP), o1);
    o2 = wmma_bf16(pa0, lds_frag_b(pV, 32, 0, AP), o2);
    o3 = wmma_bf16(pa0, lds_frag_b(pV, 48, 0, AP), o3);
    o0 = wmma_bf16(pa1, lds_frag_b(pV, 0,  32, AP), o0);
    o1 = wmma_bf16(pa1, lds_frag_b(pV, 16, 32, AP), o1);
    o2 = wmma_bf16(pa1, lds_frag_b(pV, 32, 32, AP), o2);
    o3 = wmma_bf16(pa1, lds_frag_b(pV, 48, 32, AP), o3);

    wait_async();
    __syncthreads();
    cur ^= 1;
  }

  // normalize and store merged heads [4096,1024] bf16
  const int b = bh >> 4, hIdx = bh & 15;
#pragma unroll
  for (int i = 0; i < 8; ++i) {
    float inv = 1.0f / lrow[i];
    int m = b * SEQ + q0 + w * 16 + i + 8 * hh;
    size_t base = (size_t)m * EMBED + (size_t)hIdx * DK;
    O[base + 0  + nl] = f2bf(o0[i] * inv);
    O[base + 16 + nl] = f2bf(o1[i] * inv);
    O[base + 32 + nl] = f2bf(o2[i] * inv);
    O[base + 48 + nl] = f2bf(o3[i] * inv);
  }
}

// ---------- launch ----------

extern "C" void kernel_launch(void* const* d_in, const int* in_sizes, int n_in,
                              void* d_out, int out_size, void* d_ws, size_t ws_size,
                              hipStream_t stream) {
  const float* x  = (const float*)d_in[0];
  const float* wq = (const float*)d_in[1];
  const float* wk = (const float*)d_in[2];
  const float* wv = (const float*)d_in[3];
  const float* wo = (const float*)d_in[4];
  float* out = (float*)d_out;

  u16* ws = (u16*)d_ws;
  const size_t XE = (size_t)MROWS * EMBED;  // 4M elems
  const size_t WE = (size_t)EMBED * EMBED;  // 1M elems
  u16* xb  = ws;
  u16* wqb = xb + XE;
  u16* wkb = wqb + WE;
  u16* wvb = wkb + WE;
  u16* wob = wvb + WE;
  u16* Qb  = wob + WE;
  u16* Kb  = Qb + XE;
  u16* VTb = Kb + XE;
  u16* Ob  = VTb + XE;

  f32_to_bf16_kernel<<<(int)((XE / 8 + 255) / 256), 256, 0, stream>>>(x, xb, (int)(XE / 8));
  f32_to_bf16_kernel<<<(int)((WE / 8 + 255) / 256), 256, 0, stream>>>(wq, wqb, (int)(WE / 8));
  f32_to_bf16_kernel<<<(int)((WE / 8 + 255) / 256), 256, 0, stream>>>(wk, wkb, (int)(WE / 8));
  f32_to_bf16_kernel<<<(int)((WE / 8 + 255) / 256), 256, 0, stream>>>(wv, wvb, (int)(WE / 8));
  f32_to_bf16_kernel<<<(int)((WE / 8 + 255) / 256), 256, 0, stream>>>(wo, wob, (int)(WE / 8));

  dim3 gg(MROWS / 128, EMBED / 64);
  gemm_bf16_kernel<<<gg, 256, 0, stream>>>(xb, wqb, Qb, nullptr, 0);
  gemm_bf16_kernel<<<gg, 256, 0, stream>>>(xb, wkb, Kb, nullptr, 0);
  gemm_bf16_kernel<<<gg, 256, 0, stream>>>(xb, wvb, VTb, nullptr, 2);

  attn_kernel<<<dim3(SEQ / 64, BATCH * HEADS), 128, 0, stream>>>(Qb, Kb, VTb, Ob);

  gemm_bf16_kernel<<<gg, 256, 0, stream>>>(Ob, wob, nullptr, out, 3);
}
